// PVCNN_31456340476523
// MI455X (gfx1250) — matrix operand order repaired and checked by reading, not compiled, not run on hardware
//
#include <hip/hip_runtime.h>
#include <math.h>

typedef __attribute__((ext_vector_type(16))) _Float16 v16h;
typedef __attribute__((ext_vector_type(4)))  _Float16 v4h;
typedef __attribute__((ext_vector_type(8)))  float    v8f;
typedef __attribute__((ext_vector_type(4)))  unsigned v4u;
typedef __attribute__((ext_vector_type(8)))  int      v8i;
typedef __attribute__((ext_vector_type(4)))  int      v4i;

#define BB   4
#define NP   2048
#define CHIN 22
#define CCAT 4944
// concat channel offsets
#define OFF_OH  0
#define OFF_F1  16
#define OFF_F2  80
#define OFF_F3  208
#define OFF_F4  336
#define OFF_F5  848
#define OFF_MX  2896

// A-fragment K order used here (contiguous; B tile matches): lane holds row M=lane%16;
// element e -> K = e + 16*(lane/16)
__device__ __forceinline__ int kmapA(int lane, int e) {
  return e + 16 * (lane >> 4);
}

// ---------------------------------------------------------------------------
// TDM: 1-D tile load global -> LDS (ISA 8.3/8.4 D# layout), clang-23 6-arg form
// ---------------------------------------------------------------------------
__device__ __forceinline__ void tdm_load_1d(unsigned lds_off, const void* gaddr,
                                            unsigned nbytes) {
  unsigned long long ga = (unsigned long long)gaddr;
  unsigned elems = nbytes >> 2;  // 4-byte elements
  v4u g0;
  g0.x = 1u;                                   // count=1, is_restore=0, no gather
  g0.y = lds_off;                              // lds_addr [63:32]
  g0.z = (unsigned)(ga & 0xffffffffu);         // global_addr lo
  g0.w = (unsigned)((ga >> 32) & 0x01ffffffu)  // global_addr hi (bits 120:64)
         | (2u << 30);                         // type=2 ("image")
  v8i g1;
  g1[0] = (2 << 16);                           // wg_mask=0, data_size=2 (4B)
  g1[1] = (int)((elems & 0xffffu) << 16);      // tensor_dim0[15:0] @ bits63:48
  g1[2] = (int)((elems >> 16) & 0xffffu)       // tensor_dim0[31:16]
          | (1 << 16);                         // tensor_dim1[15:0] = 1
  g1[3] = (int)((elems & 0xffffu) << 16);      // tile_dim0 @ bits127:112
  g1[4] = 1;                                   // tile_dim1=1, tile_dim2=0
  g1[5] = (int)elems;                          // tensor_dim0_stride[31:0]
  g1[6] = 0;
  g1[7] = 0;
  v4i g2 = {0, 0, 0, 0};
  v4i g3 = {0, 0, 0, 0};
  v8i g4 = {0, 0, 0, 0, 0, 0, 0, 0};
  __builtin_amdgcn_tensor_load_to_lds(g0, g1, g2, g3, g4, 0);
}

// ---------------------------------------------------------------------------
// Weight packing: GEMM A-fragments.  layout: [kc][mt][lane][e] halves
// (kc-major so one block's 16 M-tile panel per k-chunk is contiguous -> TDM)
// ---------------------------------------------------------------------------
__global__ void pack_gemm_w(const float* __restrict__ W, _Float16* __restrict__ out,
                            int M, int K, int MT, int KC) {
  size_t tot = (size_t)MT * KC * 512;
  for (size_t i = (size_t)blockIdx.x * blockDim.x + threadIdx.x; i < tot;
       i += (size_t)gridDim.x * blockDim.x) {
    int e = i & 15;
    int lane = (i >> 4) & 31;
    size_t rest = i >> 9;
    int mt = (int)(rest % MT);
    int kc = (int)(rest / MT);
    int o = mt * 16 + (lane & 15);
    int k = kc * 32 + kmapA(lane, e);
    out[i] = (o < M && k < K) ? (_Float16)W[(size_t)o * K + k] : (_Float16)0.f;
  }
}

// Conv weights [O, I, 27] -> [mt][tap(27)][kc][lane][e]
__global__ void pack_conv_w(const float* __restrict__ W, _Float16* __restrict__ out,
                            int O, int I, int MT, int KC) {
  size_t tot = (size_t)MT * 27 * KC * 512;
  for (size_t i = (size_t)blockIdx.x * blockDim.x + threadIdx.x; i < tot;
       i += (size_t)gridDim.x * blockDim.x) {
    int e = i & 15;
    int lane = (i >> 4) & 31;
    size_t rest = i >> 9;
    int kc = (int)(rest % KC);
    rest /= KC;
    int tap = (int)(rest % 27);
    int mt = (int)(rest / 27);
    int o = mt * 16 + (lane & 15);
    int ci = kc * 32 + kmapA(lane, e);
    out[i] = (o < O && ci < I) ? (_Float16)W[((size_t)o * I + ci) * 27 + tap]
                               : (_Float16)0.f;
  }
}

// ---------------------------------------------------------------------------
// WMMA GEMM:  Y[b,o,n] = act( s[o] * sum_k W[o,k] X[b,k,n] + t[o] )
// grid: (NP/16, ceil(MT/16), BB); block 128.  Each wave: 4 accumulators
// (4 M-tiles), so a block covers 256 output rows per staged B tile.
// A panels stream through LDS via the Tensor Data Mover.
// act: 0 = none, 1 = relu
// ---------------------------------------------------------------------------
__global__ void __launch_bounds__(128) gemm_wmma(
    const _Float16* __restrict__ Apack,
    const float* __restrict__ X, size_t sXb,
    float* __restrict__ Y, size_t sYb,
    const float* __restrict__ s, const float* __restrict__ t,
    int M, int K, int KC, int MT, int act) {
  const int b = blockIdx.z;
  const int n0 = blockIdx.x * 16;
  const int lane = threadIdx.x & 31;
  const int wv = threadIdx.x >> 5;
  const int mt0 = blockIdx.y * 16;
  const int tiles = min(16, MT - mt0);

  __shared__ __align__(64) _Float16 bt[16][32];     // B tile  [col][k]
  __shared__ __align__(64) _Float16 apan[16 * 512]; // A panel: 16 Mtiles x 1KB

  v8f acc[4];
#pragma unroll
  for (int j = 0; j < 4; ++j) acc[j] = (v8f){0.f,0.f,0.f,0.f,0.f,0.f,0.f,0.f};

  const float* Xb = X + (size_t)b * sXb;
  const int col = threadIdx.x & 15;
  const int kg = threadIdx.x >> 4;  // 0..7, each owns 4 consecutive k rows

  for (int kc = 0; kc < KC; ++kc) {
    __syncthreads();
    // ---- stage B tile: 4 clamped loads + one ds_store_b64, no branches ----
    {
      v4h hv;
#pragma unroll
      for (int j = 0; j < 4; ++j) {
        int kk = kc * 32 + kg * 4 + j;
        int kkc = min(kk, K - 1);
        float v = Xb[(size_t)kkc * NP + n0 + col];
        hv[j] = (kk < K) ? (_Float16)v : (_Float16)0.f;
      }
      *(v4h*)&bt[col][kg * 4] = hv;
    }
    // ---- TDM: stream this k-chunk's A panel into LDS (wave 0 issues) ----
    if (wv == 0) {
      tdm_load_1d((unsigned)(size_t)&apan[0],
                  Apack + ((size_t)kc * MT + mt0) * 512,
                  (unsigned)tiles * 1024u);
      __builtin_amdgcn_s_wait_tensorcnt(0);
    }
    __syncthreads();

    v16h bf = *(const v16h*)&bt[lane & 15][(lane >> 4) * 16];
#pragma unroll
    for (int j = 0; j < 4; ++j) {
      int ml = min(wv * 4 + j, tiles - 1);
      v16h af = *(const v16h*)&apan[((size_t)ml * 32 + lane) * 16];
      acc[j] = __builtin_amdgcn_wmma_f32_16x16x32_f16(false, af, false, bf,
                                                      (short)0, acc[j], false, false);
    }
  }

  const int half = lane >> 4;
#pragma unroll
  for (int j = 0; j < 4; ++j) {
    int mt = mt0 + wv * 4 + j;
#pragma unroll
    for (int g = 0; g < 8; ++g) {
      int o = mt * 16 + g + 8 * half;
      if (o < M) {
        float yv = acc[j][g] * (s ? s[o] : 1.f) + (t ? t[o] : 0.f);
        if (act == 1) yv = fmaxf(yv, 0.f);
        Y[(size_t)b * sYb + (size_t)o * NP + n0 + (lane & 15)] = yv;
      }
    }
  }
}

// ---------------------------------------------------------------------------
// Implicit-GEMM 3x3x3 conv (SAME, stride 1) + scale/bias + leaky-relu(0.1)
// grid: (r*r*(r/16), ceil(MT/8), BB); block 128; 2 accumulators per wave
// so one block covers up to 128 output channels (grid read exactly once).
// ---------------------------------------------------------------------------
__global__ void __launch_bounds__(128) conv3d_wmma(
    const _Float16* __restrict__ Apack,
    const float* __restrict__ G, float* __restrict__ O,
    const float* __restrict__ s, const float* __restrict__ t,
    int Cin, int KC, int Cout, int MT, int r) {
  const int r2 = r * r, r3 = r2 * r;
  const int ZT = r >> 4;
  const int b = blockIdx.z;
  const int zt = blockIdx.x % ZT;
  const int xy = blockIdx.x / ZT;
  const int x = xy / r, y = xy % r;
  const int z0 = zt * 16;
  const int lane = threadIdx.x & 31;
  const int wv = threadIdx.x >> 5;
  const int mtb = blockIdx.y * 8 + wv * 2;

  __shared__ __align__(64) _Float16 bt[16][32];
  v8f acc[2];
  acc[0] = (v8f){0.f,0.f,0.f,0.f,0.f,0.f,0.f,0.f};
  acc[1] = acc[0];
  const float* Gb = G + (size_t)b * Cin * r3;
  const int col = threadIdx.x & 15;
  const int kg = threadIdx.x >> 4;

  for (int tap = 0; tap < 27; ++tap) {
    int dx = tap / 9 - 1, dy = (tap / 3) % 3 - 1, dz = tap % 3 - 1;
    int xx = x + dx, yy = y + dy;
    if (xx < 0 || xx >= r || yy < 0 || yy >= r) continue;  // uniform per block
    const float* Grow = Gb + (size_t)(xx * r + yy) * r;
    int zz = z0 + col + dz;
    int zzc = min(max(zz, 0), r - 1);
    bool zin = (zz >= 0) && (zz < r);
    for (int kc = 0; kc < KC; ++kc) {
      __syncthreads();
      {
        v4h hv;
#pragma unroll
        for (int j = 0; j < 4; ++j) {
          int ci = kc * 32 + kg * 4 + j;
          int cic = min(ci, Cin - 1);
          float v = Grow[(size_t)cic * r3 + zzc];
          hv[j] = (zin && ci < Cin) ? (_Float16)v : (_Float16)0.f;
        }
        *(v4h*)&bt[col][kg * 4] = hv;
      }
      __syncthreads();
      v16h bf = *(const v16h*)&bt[lane & 15][(lane >> 4) * 16];
#pragma unroll
      for (int j = 0; j < 2; ++j) {
        int mt = min(mtb + j, MT - 1);
        v16h af = *(const v16h*)(Apack +
                   (((((size_t)mt * 27 + tap) * KC + kc) * 32 + lane) << 4));
        acc[j] = __builtin_amdgcn_wmma_f32_16x16x32_f16(false, af, false, bf,
                                                        (short)0, acc[j], false, false);
      }
    }
  }

  const int half = lane >> 4;
  const size_t vb = (size_t)(x * r + y) * r + z0 + (lane & 15);
#pragma unroll
  for (int j = 0; j < 2; ++j) {
#pragma unroll
    for (int g = 0; g < 8; ++g) {
      int o = (mtb + j) * 16 + g + 8 * half;
      if (o < Cout) {
        float yv = acc[j][g] * s[o] + t[o];
        yv = (yv >= 0.f) ? yv : 0.1f * yv;
        O[((size_t)b * Cout + o) * r3 + vb] = yv;
      }
    }
  }
}

// ---------------------------------------------------------------------------
// Small helper kernels
// ---------------------------------------------------------------------------
__global__ void fill_f32(float* __restrict__ p, float v, size_t n) {
  for (size_t i = (size_t)blockIdx.x * blockDim.x + threadIdx.x; i < n;
       i += (size_t)gridDim.x * blockDim.x)
    p[i] = v;
}

__global__ void coords_mean(const float* __restrict__ inp, float* __restrict__ cmean) {
  int b = blockIdx.x;
  __shared__ float red[256];
  for (int a = 0; a < 3; ++a) {
    float sm = 0.f;
    for (int n = threadIdx.x; n < NP; n += 256)
      sm += inp[((size_t)b * CHIN + a) * NP + n];
    red[threadIdx.x] = sm;
    __syncthreads();
    for (int st = 128; st > 0; st >>= 1) {
      if (threadIdx.x < st) red[threadIdx.x] += red[threadIdx.x + st];
      __syncthreads();
    }
    if (threadIdx.x == 0) cmean[b * 3 + a] = red[0] / (float)NP;
    __syncthreads();
  }
}

__device__ __forceinline__ float normc(float c, float m, int r) {
  float v = (c - m + 1.0f) * (0.5f * (float)r);
  return fminf(fmaxf(v, 0.f), (float)(r - 1));
}

__global__ void voxel_scatter(const float* __restrict__ F, size_t sFb,
                              const float* __restrict__ inp,
                              const float* __restrict__ cmean,
                              float* __restrict__ grid, float* __restrict__ cnt,
                              int C, int r) {
  int gid = blockIdx.x * blockDim.x + threadIdx.x;
  if (gid >= BB * NP) return;
  int b = gid / NP, n = gid % NP;
  int r3 = r * r * r;
  int vx = (int)rintf(normc(inp[((size_t)b * CHIN + 0) * NP + n], cmean[b * 3 + 0], r));
  int vy = (int)rintf(normc(inp[((size_t)b * CHIN + 1) * NP + n], cmean[b * 3 + 1], r));
  int vz = (int)rintf(normc(inp[((size_t)b * CHIN + 2) * NP + n], cmean[b * 3 + 2], r));
  int idx = (vx * r + vy) * r + vz;
  atomicAdd(&cnt[(size_t)b * r3 + idx], 1.0f);
  for (int c = 0; c < C; ++c)
    atomicAdd(&grid[((size_t)b * C + c) * r3 + idx], F[(size_t)b * sFb + (size_t)c * NP + n]);
}

__global__ void voxel_norm(float* __restrict__ grid, const float* __restrict__ cnt,
                           int C, int r3) {
  size_t tot = (size_t)BB * C * r3;
  for (size_t i = (size_t)blockIdx.x * blockDim.x + threadIdx.x; i < tot;
       i += (size_t)gridDim.x * blockDim.x) {
    int b = (int)(i / ((size_t)C * r3));
    int v = (int)(i % r3);
    grid[i] = grid[i] / fmaxf(cnt[(size_t)b * r3 + v], 1.f);
  }
}

__global__ void grid_mean(const float* __restrict__ grid, float* __restrict__ z0, int r3) {
  int bc = blockIdx.x;
  __shared__ float red[256];
  float sm = 0.f;
  const float* g = grid + (size_t)bc * r3;
  for (int v = threadIdx.x; v < r3; v += 256) sm += g[v];
  red[threadIdx.x] = sm;
  __syncthreads();
  for (int st = 128; st > 0; st >>= 1) {
    if (threadIdx.x < st) red[threadIdx.x] += red[threadIdx.x + st];
    __syncthreads();
  }
  if (threadIdx.x == 0) z0[bc] = red[0] / (float)r3;
}

__global__ void se_fc(const float* __restrict__ z0,
                      const float* __restrict__ w1, const float* __restrict__ b1,
                      const float* __restrict__ w2, const float* __restrict__ b2,
                      float* __restrict__ z, int C) {
  int b = blockIdx.x;
  int H = C / 8;
  __shared__ float h[16];
  int tid = threadIdx.x;
  if (tid < H) {
    float a = b1[tid];
    for (int c = 0; c < C; ++c) a += w1[(size_t)tid * C + c] * z0[(size_t)b * C + c];
    h[tid] = fmaxf(a, 0.f);
  }
  __syncthreads();
  if (tid < C) {
    float a = b2[tid];
    for (int j = 0; j < H; ++j) a += w2[(size_t)tid * H + j] * h[j];
    z[(size_t)b * C + tid] = 1.f / (1.f + expf(-a));
  }
}

__global__ void devox_add(const float* __restrict__ G, const float* __restrict__ z,
                          const float* __restrict__ inp, const float* __restrict__ cmean,
                          float* __restrict__ out, size_t sOb, int C, int r) {
  size_t tot = (size_t)BB * C * NP;
  size_t gid = (size_t)blockIdx.x * blockDim.x + threadIdx.x;
  if (gid >= tot) return;
  int n = (int)(gid % NP);
  int c = (int)((gid / NP) % C);
  int b = (int)(gid / ((size_t)NP * C));
  float nx = normc(inp[((size_t)b * CHIN + 0) * NP + n], cmean[b * 3 + 0], r);
  float ny = normc(inp[((size_t)b * CHIN + 1) * NP + n], cmean[b * 3 + 1], r);
  float nz = normc(inp[((size_t)b * CHIN + 2) * NP + n], cmean[b * 3 + 2], r);
  int lx = (int)floorf(nx), ly = (int)floorf(ny), lz = (int)floorf(nz);
  int hx = min(lx + 1, r - 1), hy = min(ly + 1, r - 1), hz = min(lz + 1, r - 1);
  float dx = nx - (float)lx, dy = ny - (float)ly, dz = nz - (float)lz;
  const float* Gc = G + ((size_t)b * C + c) * ((size_t)r * r * r);
  float acc = 0.f;
#pragma unroll
  for (int sx = 0; sx < 2; ++sx)
#pragma unroll
    for (int sy = 0; sy < 2; ++sy)
#pragma unroll
      for (int sz = 0; sz < 2; ++sz) {
        int ix = sx ? hx : lx, iy = sy ? hy : ly, iz = sz ? hz : lz;
        float w = (sx ? dx : 1.f - dx) * (sy ? dy : 1.f - dy) * (sz ? dz : 1.f - dz);
        acc += w * Gc[((size_t)ix * r + iy) * r + iz];
      }
  out[(size_t)b * sOb + (size_t)c * NP + n] += z[(size_t)b * C + c] * acc;
}

__global__ void copy_onehot(const float* __restrict__ inp, float* __restrict__ cc) {
  size_t tot = (size_t)BB * 16 * NP;
  size_t gid = (size_t)blockIdx.x * blockDim.x + threadIdx.x;
  if (gid >= tot) return;
  int n = (int)(gid % NP);
  int c = (int)((gid / NP) % 16);
  int b = (int)(gid / ((size_t)NP * 16));
  cc[(size_t)b * CCAT * NP + (size_t)(OFF_OH + c) * NP + n] =
      inp[((size_t)b * CHIN + 6 + c) * NP + n];
}

__global__ void rowmax(const float* __restrict__ X, size_t sXb, int C,
                       float* __restrict__ out) {
  int bc = blockIdx.x;
  int b = bc / C, c = bc % C;
  __shared__ float red[256];
  float m = -INFINITY;
  const float* row = X + (size_t)b * sXb + (size_t)c * NP;
  for (int n = threadIdx.x; n < NP; n += 256) m = fmaxf(m, row[n]);
  red[threadIdx.x] = m;
  __syncthreads();
  for (int st = 128; st > 0; st >>= 1) {
    if (threadIdx.x < st) red[threadIdx.x] = fmaxf(red[threadIdx.x], red[threadIdx.x + st]);
    __syncthreads();
  }
  if (threadIdx.x == 0) out[bc] = red[0];
}

__global__ void bcast_rows(const float* __restrict__ v, float* __restrict__ out,
                           size_t sOb, int C) {
  size_t tot = (size_t)BB * C * NP;
  size_t gid = (size_t)blockIdx.x * blockDim.x + threadIdx.x;
  if (gid >= tot) return;
  int n = (int)(gid % NP);
  int c = (int)((gid / NP) % C);
  int b = (int)(gid / ((size_t)NP * C));
  out[(size_t)b * sOb + (size_t)c * NP + n] = v[(size_t)b * C + c];
}

// ---------------------------------------------------------------------------
// Host
// ---------------------------------------------------------------------------
extern "C" void kernel_launch(void* const* d_in, const int* in_sizes, int n_in,
                              void* d_out, int out_size, void* d_ws, size_t ws_size,
                              hipStream_t stream) {
  (void)in_sizes; (void)n_in; (void)out_size; (void)ws_size;
  auto F = [&](int i) { return (const float*)d_in[i]; };
  const float* inp = F(0);
  // sorted-key pytree flatten order
  const float *cls1_s = F(1), *cls1_t = F(2), *cls1_w = F(3);
  const float *cls2_s = F(4), *cls2_t = F(5), *cls2_w = F(6);
  const float *cls3_s = F(7), *cls3_t = F(8), *cls3_w = F(9);
  const float *clsb = F(10), *clsw = F(11);
  const float *m4_s = F(12), *m4_t = F(13), *m4_w = F(14);
  const float *m5_s = F(15), *m5_t = F(16), *m5_w = F(17);
  struct PV {
    const float *ps, *pt, *pw, *s1, *s2, *seb1, *seb2, *sew1, *sew2, *t1, *t2, *w1, *w2;
  };
  PV pv[3];
  for (int i = 0; i < 3; ++i) {
    int o = 18 + 13 * i;
    pv[i] = {F(o+0), F(o+1), F(o+2), F(o+3), F(o+4), F(o+5), F(o+6),
             F(o+7), F(o+8), F(o+9), F(o+10), F(o+11), F(o+12)};
  }

  // workspace carving
  char* base = (char*)d_ws;
  size_t off = 0;
  auto carve = [&](size_t bytes) -> void* {
    off = (off + 255) & ~(size_t)255;
    void* q = base + off;
    off += bytes;
    return q;
  };
  float* cmean = (float*)carve(BB * 3 * sizeof(float));
  float* cnt   = (float*)carve((size_t)BB * 32768 * sizeof(float));
  float* gridA = (float*)carve((size_t)BB * 64 * 32768 * sizeof(float));
  float* gridB = (float*)carve((size_t)BB * 64 * 32768 * sizeof(float));
  float* cc    = (float*)carve((size_t)BB * CCAT * NP * sizeof(float));
  float* z0m   = (float*)carve(BB * 128 * sizeof(float));
  float* zse   = (float*)carve(BB * 128 * sizeof(float));
  float* fmx   = (float*)carve(BB * 2048 * sizeof(float));
  float* x1    = (float*)carve((size_t)BB * 256 * NP * sizeof(float));
  float* x2    = (float*)carve((size_t)BB * 256 * NP * sizeof(float));
  float* x3    = (float*)carve((size_t)BB * 128 * NP * sizeof(float));
  auto carveH = [&](size_t halves) { return (_Float16*)carve(halves * sizeof(_Float16)); };

  auto MTof = [](int M) { return (M + 15) / 16; };   // exact 16-row tile count
  auto KCof = [](int K) { return (K + 31) / 32; };

  // packed weight buffers
  _Float16 *pk_w1[3], *pk_w2[3], *pk_pw[3];
  int cin[3] = {6, 64, 128}, cout[3] = {64, 128, 128}, rr[3] = {32, 16, 16};
  for (int i = 0; i < 3; ++i) {
    pk_w1[i] = carveH((size_t)MTof(cout[i]) * 27 * KCof(cin[i]) * 512);
    pk_w2[i] = carveH((size_t)MTof(cout[i]) * 27 * KCof(cout[i]) * 512);
    pk_pw[i] = carveH((size_t)MTof(cout[i]) * KCof(cin[i]) * 512);
  }
  _Float16* pk_m4 = carveH((size_t)MTof(512) * KCof(128) * 512);
  _Float16* pk_m5 = carveH((size_t)MTof(2048) * KCof(512) * 512);
  _Float16* pk_c1 = carveH((size_t)MTof(256) * KCof(CCAT) * 512);
  _Float16* pk_c2 = carveH((size_t)MTof(256) * KCof(256) * 512);
  _Float16* pk_c3 = carveH((size_t)MTof(128) * KCof(256) * 512);
  _Float16* pk_cw = carveH((size_t)MTof(50) * KCof(128) * 512);

  auto packG = [&](const float* W, _Float16* out, int M, int K) {
    size_t tot = (size_t)MTof(M) * KCof(K) * 512;
    pack_gemm_w<<<(unsigned)((tot + 255) / 256), 256, 0, stream>>>(W, out, M, K, MTof(M), KCof(K));
  };
  auto packC = [&](const float* W, _Float16* out, int O, int I) {
    size_t tot = (size_t)MTof(O) * 27 * KCof(I) * 512;
    pack_conv_w<<<(unsigned)((tot + 255) / 256), 256, 0, stream>>>(W, out, O, I, MTof(O), KCof(I));
  };
  for (int i = 0; i < 3; ++i) {
    packC(pv[i].w1, pk_w1[i], cout[i], cin[i]);
    packC(pv[i].w2, pk_w2[i], cout[i], cout[i]);
    packG(pv[i].pw, pk_pw[i], cout[i], cin[i]);
  }
  packG(m4_w, pk_m4, 512, 128);
  packG(m5_w, pk_m5, 2048, 512);
  packG(cls1_w, pk_c1, 256, CCAT);
  packG(cls2_w, pk_c2, 256, 256);
  packG(cls3_w, pk_c3, 128, 256);
  packG(clsw, pk_cw, 50, 128);

  auto gemm = [&](const _Float16* A, const float* X, size_t sXb, float* Y, size_t sYb,
                  const float* s, const float* t, int M, int K, int act) {
    int MT = MTof(M);
    dim3 g(NP / 16, (MT + 15) / 16, BB);
    gemm_wmma<<<g, 128, 0, stream>>>(A, X, sXb, Y, sYb, s, t, M, K, KCof(K), MT, act);
  };
  auto fill = [&](float* p, float v, size_t n) {
    fill_f32<<<(unsigned)((n + 255) / 256), 256, 0, stream>>>(p, v, n);
  };

  coords_mean<<<BB, 256, 0, stream>>>(inp, cmean);
  {
    size_t tot = (size_t)BB * 16 * NP;
    copy_onehot<<<(unsigned)((tot + 255) / 256), 256, 0, stream>>>(inp, cc);
  }

  const float* fin = inp;          // pv1 input features = inputs[:, :6, :]
  size_t sFb = (size_t)CHIN * NP;
  int foff[3] = {OFF_F1, OFF_F2, OFF_F3};
  for (int i = 0; i < 3; ++i) {
    int r = rr[i], Ci = cin[i], Co = cout[i];
    int r3 = r * r * r;
    int MT = MTof(Co);
    float* fout = cc + (size_t)foff[i] * NP;  // [b] stride CCAT*NP, channel stride NP
    fill(gridA, 0.f, (size_t)BB * Ci * r3);
    fill(cnt, 0.f, (size_t)BB * r3);
    voxel_scatter<<<(BB * NP + 255) / 256, 256, 0, stream>>>(fin, sFb, inp, cmean,
                                                             gridA, cnt, Ci, r);
    {
      size_t tot = (size_t)BB * Ci * r3;
      voxel_norm<<<(unsigned)((tot + 255) / 256), 256, 0, stream>>>(gridA, cnt, Ci, r3);
    }
    dim3 gc(r * r * (r / 16), (MT + 7) / 8, BB);
    conv3d_wmma<<<gc, 128, 0, stream>>>(pk_w1[i], gridA, gridB, pv[i].s1, pv[i].t1,
                                        Ci, KCof(Ci), Co, MT, r);
    conv3d_wmma<<<gc, 128, 0, stream>>>(pk_w2[i], gridB, gridA, pv[i].s2, pv[i].t2,
                                        Co, KCof(Co), Co, MT, r);
    grid_mean<<<BB * Co, 256, 0, stream>>>(gridA, z0m, r3);
    se_fc<<<BB, 128, 0, stream>>>(z0m, pv[i].sew1, pv[i].seb1, pv[i].sew2, pv[i].seb2,
                                  zse, Co);
    // point branch: relu(ps * (pw @ f) + pt) -> fout
    gemm(pk_pw[i], fin, sFb, fout, (size_t)CCAT * NP, pv[i].ps, pv[i].pt, Co, Ci, 1);
    {
      size_t tot = (size_t)BB * Co * NP;
      devox_add<<<(unsigned)((tot + 255) / 256), 256, 0, stream>>>(
          gridA, zse, inp, cmean, fout, (size_t)CCAT * NP, Co, r);
    }
    fin = fout;
    sFb = (size_t)CCAT * NP;
  }

  // mlp4: f3(128) -> f4(512), mlp5: f4 -> f5(2048)
  gemm(pk_m4, cc + (size_t)OFF_F3 * NP, (size_t)CCAT * NP,
       cc + (size_t)OFF_F4 * NP, (size_t)CCAT * NP, m4_s, m4_t, 512, 128, 1);
  gemm(pk_m5, cc + (size_t)OFF_F4 * NP, (size_t)CCAT * NP,
       cc + (size_t)OFF_F5 * NP, (size_t)CCAT * NP, m5_s, m5_t, 2048, 512, 1);
  // global max over points of f5, broadcast into concat
  rowmax<<<BB * 2048, 256, 0, stream>>>(cc + (size_t)OFF_F5 * NP, (size_t)CCAT * NP,
                                        2048, fmx);
  {
    size_t tot = (size_t)BB * 2048 * NP;
    bcast_rows<<<(unsigned)((tot + 255) / 256), 256, 0, stream>>>(
        fmx, cc + (size_t)OFF_MX * NP, (size_t)CCAT * NP, 2048);
  }

  // classifier head
  gemm(pk_c1, cc, (size_t)CCAT * NP, x1, (size_t)256 * NP, cls1_s, cls1_t, 256, CCAT, 1);
  gemm(pk_c2, x1, (size_t)256 * NP, x2, (size_t)256 * NP, cls2_s, cls2_t, 256, 256, 1);
  gemm(pk_c3, x2, (size_t)256 * NP, x3, (size_t)128 * NP, cls3_s, cls3_t, 128, 256, 1);
  gemm(pk_cw, x3, (size_t)128 * NP, (float*)d_out, (size_t)50 * NP,
       nullptr, clsb, 50, 128, 0);
}
